// GCN_41231686041975
// MI455X (gfx1250) — matrix operand order repaired and checked by reading, not compiled, not run on hardware
//
#include <hip/hip_runtime.h>
#include <stdint.h>

typedef __attribute__((ext_vector_type(16))) __bf16 v16bf;
typedef __attribute__((ext_vector_type(8)))  __bf16 v8bf;
typedef __attribute__((ext_vector_type(8)))  float  v8f;

__device__ __forceinline__ unsigned short f2bf(float f) {
    unsigned int u = __float_as_uint(f);
    unsigned int r = u + 0x7FFFu + ((u >> 16) & 1u);   // round to nearest even
    return (unsigned short)(r >> 16);
}

// ---------------- elementwise / graph-normalization kernels ----------------

__global__ void k_zero(float* __restrict__ p, int n) {
    int i = blockIdx.x * blockDim.x + threadIdx.x;
    if (i < n) p[i] = 0.0f;
}

__global__ void k_deg(const float* __restrict__ ew, const int* __restrict__ src,
                      float* __restrict__ deg, int E) {
    int e = blockIdx.x * blockDim.x + threadIdx.x;
    if (e < E) atomicAdd(deg + src[e], ew[e]);
}

__global__ void k_dis(float* __restrict__ deg, int n) {
    int i = blockIdx.x * blockDim.x + threadIdx.x;
    if (i < n) {
        float d = deg[i];
        deg[i] = (d > 0.0f) ? rsqrtf(d) : 0.0f;
    }
}

__global__ void k_normw(const int* __restrict__ src, const int* __restrict__ dst,
                        const float* __restrict__ ew, const float* __restrict__ dis,
                        float* __restrict__ nw, int E) {
    int e = blockIdx.x * blockDim.x + threadIdx.x;
    if (e < E) nw[e] = -dis[src[e]] * ew[e] * dis[dst[e]];
}

// Repack fp32 weight (Kt x Nout, row-major) -> bf16 transposed (Nout x Kt)
__global__ void k_repack(const float* __restrict__ W, unsigned short* __restrict__ Wt,
                         int Kt, int Nout) {
    int i = blockIdx.x * blockDim.x + threadIdx.x;
    if (i < Kt * Nout) {
        int k = i / Nout, n = i % Nout;
        Wt[(size_t)n * Kt + k] = f2bf(W[i]);
    }
}

// Convert fp32 (rows x C, dense) into bf16 slot of a wider row (lda), at colOff
__global__ void k_cvt(const float* __restrict__ H, int C,
                      unsigned short* __restrict__ out, int lda, int colOff, int total) {
    int i = blockIdx.x * blockDim.x + threadIdx.x;
    if (i < total) {
        int r = i / C, c = i % C;
        out[(size_t)r * lda + colOff + c] = f2bf(H[i]);
    }
}

// T2 = 2*T2 - X
__global__ void k_combine(float* __restrict__ T2, const float* __restrict__ X, int n) {
    int i = blockIdx.x * blockDim.x + threadIdx.x;
    if (i < n) T2[i] = 2.0f * T2[i] - X[i];
}

// out[dst] += nw[e] * h[src] ; one wave (32 lanes) per edge, float4 per lane (128 feats)
__global__ void k_scatter(const float* __restrict__ h,
                          const int* __restrict__ src, const int* __restrict__ dst,
                          const float* __restrict__ nw, float* __restrict__ out, int E) {
    int e = blockIdx.x * (blockDim.x >> 5) + (threadIdx.x >> 5);
    if (e >= E) return;
    int lane = threadIdx.x & 31;
    int s = src[e], d = dst[e];
    float w = nw[e];
    const float4 hv = *(const float4*)(h + (size_t)s * 128 + lane * 4);
    float* o = out + (size_t)d * 128 + lane * 4;
    atomicAdd(o + 0, w * hv.x);
    atomicAdd(o + 1, w * hv.y);
    atomicAdd(o + 2, w * hv.z);
    atomicAdd(o + 3, w * hv.w);
}

// ---------------- WMMA GEMM: C = relu(A(bf16) @ B + bias), fp32 accumulate ---------
// A: M x Kt bf16 row-major (lda), Bt: Nout x Kt bf16 (B transposed, ldb=Kt)
// Block = 256 threads = 8 waves; each wave -> 16x32 output tile; block -> 128x32.
__global__ __launch_bounds__(256)
void k_gemm(const unsigned short* __restrict__ Au, int lda,
            const unsigned short* __restrict__ Btu, int ldb,
            const float* __restrict__ bias,
            float* __restrict__ C, int ldc,
            int M, int Kt) {
    const int wave = threadIdx.x >> 5;
    const int lane = threadIdx.x & 31;
    const int half = lane >> 4;          // 0: lanes 0-15, 1: lanes 16-31
    const int l15  = lane & 15;
    const int mBase = blockIdx.x * 128 + wave * 16;
    const int nBase = blockIdx.y * 32;

    const __bf16* A  = (const __bf16*)Au;
    const __bf16* Bt = (const __bf16*)Btu;

    // A fragment: row = mBase+l15; this lane holds K = ks0+[0..15] contiguous
    const int ks0 = half ? 16 : 0;
    const __bf16* aRow = A + (size_t)(mBase + l15) * lda + ks0;
    // B fragment: col = nBase+l15 (+16 for 2nd tile); K = kOffB+[0..15] contiguous in Bt
    const int kOffB = half ? 16 : 0;
    const __bf16* b0p = Bt + (size_t)(nBase + l15) * ldb + kOffB;
    const __bf16* b1p = b0p + (size_t)16 * ldb;

    v8f c0 = {};
    v8f c1 = {};
    for (int k = 0; k < Kt; k += 32) {
        v8bf alo = *(const v8bf*)(aRow + k);
        v8bf ahi = *(const v8bf*)(aRow + k + 8);
        v16bf a;
#pragma unroll
        for (int i = 0; i < 8; ++i) { a[i] = alo[i]; a[i + 8] = ahi[i]; }
        v16bf b0 = *(const v16bf*)(b0p + k);
        v16bf b1 = *(const v16bf*)(b1p + k);
        c0 = __builtin_amdgcn_wmma_f32_16x16x32_bf16(false, a, false, b0, (short)0, c0, false, false);
        c1 = __builtin_amdgcn_wmma_f32_16x16x32_bf16(false, a, false, b1, (short)0, c1, false, false);
    }

    const float bias0 = bias[nBase + l15];
    const float bias1 = bias[nBase + 16 + l15];
    const int rBase = mBase + (half ? 8 : 0);   // C layout: VGPR r -> row r (+8 for hi half)
#pragma unroll
    for (int r = 0; r < 8; ++r) {
        int rr = rBase + r;
        if (rr < M) {
            float v0 = c0[r] + bias0; v0 = v0 > 0.0f ? v0 : 0.0f;
            float v1 = c1[r] + bias1; v1 = v1 > 0.0f ? v1 : 0.0f;
            C[(size_t)rr * ldc + nBase + l15]      = v0;
            C[(size_t)rr * ldc + nBase + 16 + l15] = v1;
        }
    }
}

// ---------------------------------------------------------------------------

extern "C" void kernel_launch(void* const* d_in, const int* in_sizes, int n_in,
                              void* d_out, int out_size, void* d_ws, size_t ws_size,
                              hipStream_t stream) {
    const float* x   = (const float*)d_in[0];
    const int*   ei  = (const int*)d_in[1];
    const float* ew  = (const float*)d_in[2];
    // d_in[3] = dp (0, unused)
    const float* W1  = (const float*)d_in[4];
    const float* b1  = (const float*)d_in[5];
    const float* W2  = (const float*)d_in[6];
    const float* b2  = (const float*)d_in[7];
    const float* Wl1 = (const float*)d_in[8];
    const float* bl1 = (const float*)d_in[9];
    const float* Wl2 = (const float*)d_in[10];
    const float* bl2 = (const float*)d_in[11];
    const float* Wo  = (const float*)d_in[12];
    const float* bo  = (const float*)d_in[13];

    const int FDIM = 128;
    const int N = in_sizes[0] / FDIM;
    const int E = in_sizes[2];
    const int* src = ei;
    const int* dst = ei + E;
    const int Mp = ((N + 127) / 128) * 128;   // pad M for 128-row GEMM blocks

    // ---- bump allocator over d_ws ----
    char* wsb = (char*)d_ws;
    size_t off = 0;
    auto alloc = [&](size_t bytes) -> void* {
        void* p = (void*)(wsb + off);
        off += (bytes + 255) & ~(size_t)255;
        return p;
    };
    float* dis   = (float*)alloc((size_t)N * 4);              // deg, then 1/sqrt(deg)
    float* nw    = (float*)alloc((size_t)E * 4);
    float* T1    = (float*)alloc((size_t)Mp * 128 * 4);       // T1,T2 contiguous ->
    float* T2    = (float*)alloc((size_t)Mp * 128 * 4);       //   reused as Mp x 256 f32
    unsigned short* Acat = (unsigned short*)alloc((size_t)Mp * 384 * 2);
    float* H1f   = (float*)alloc((size_t)Mp * 128 * 4);
    float* H2f   = (float*)alloc((size_t)Mp * 128 * 4);
    unsigned short* Hb   = (unsigned short*)alloc((size_t)Mp * 256 * 2);
    unsigned short* W1t  = (unsigned short*)alloc((size_t)384 * 128 * 2);
    unsigned short* W2t  = (unsigned short*)alloc((size_t)384 * 128 * 2);
    unsigned short* Wl1t = (unsigned short*)alloc((size_t)128 * 256 * 2);
    unsigned short* Wl2t = (unsigned short*)alloc((size_t)256 * 128 * 2);
    unsigned short* Wot  = (unsigned short*)alloc((size_t)128 * 64 * 2);
    float* D1 = T1;                                           // Mp x 256 f32 (dense-1 out)

    const int TB = 256;
    auto nb = [&](long n) { return (int)((n + TB - 1) / TB); };

    // ---- scaled-Laplacian edge weights ----
    k_zero <<<nb(N), TB, 0, stream>>>(dis, N);
    k_deg  <<<nb(E), TB, 0, stream>>>(ew, src, dis, E);
    k_dis  <<<nb(N), TB, 0, stream>>>(dis, N);
    k_normw<<<nb(E), TB, 0, stream>>>(src, dst, ew, dis, nw, E);

    // ---- weight repack to bf16, transposed (Nout x Kt) ----
    k_repack<<<nb(384 * 128), TB, 0, stream>>>(W1,  W1t,  384, 128);
    k_repack<<<nb(384 * 128), TB, 0, stream>>>(W2,  W2t,  384, 128);
    k_repack<<<nb(128 * 256), TB, 0, stream>>>(Wl1, Wl1t, 128, 256);
    k_repack<<<nb(256 * 128), TB, 0, stream>>>(Wl2, Wl2t, 256, 128);
    k_repack<<<nb(128 * 64),  TB, 0, stream>>>(Wo,  Wot,  128, 64);

    const int NC = N * 128;
    const int scatterBlocks = (E + 7) / 8;     // 8 edges (waves) per 256-thread block

    // ---- one ChebConv layer: relu([X|prop(X)|2*prop(prop(X))-X] @ Wcat + b) ----
    auto cheb = [&](const float* Xf, const unsigned short* Wt, const float* bias, float* Hout) {
        k_cvt    <<<nb(NC), TB, 0, stream>>>(Xf, 128, Acat, 384, 0, NC);       // Tx0
        k_zero   <<<nb(NC), TB, 0, stream>>>(T1, NC);
        k_scatter<<<scatterBlocks, 256, 0, stream>>>(Xf, src, dst, nw, T1, E); // Tx1
        k_cvt    <<<nb(NC), TB, 0, stream>>>(T1, 128, Acat, 384, 128, NC);
        k_zero   <<<nb(NC), TB, 0, stream>>>(T2, NC);
        k_scatter<<<scatterBlocks, 256, 0, stream>>>(T1, src, dst, nw, T2, E);
        k_combine<<<nb(NC), TB, 0, stream>>>(T2, Xf, NC);                      // Tx2
        k_cvt    <<<nb(NC), TB, 0, stream>>>(T2, 128, Acat, 384, 256, NC);
        dim3 g(Mp / 128, 128 / 32);
        k_gemm<<<g, 256, 0, stream>>>(Acat, 384, Wt, 384, bias, Hout, 128, N, 384);
    };

    cheb(x,   W1t, b1, H1f);
    cheb(H1f, W2t, b2, H2f);

    // ---- MLP head: 128 -> 256 -> 128 -> 64, all relu ----
    k_cvt<<<nb(NC), TB, 0, stream>>>(H2f, 128, Hb, 128, 0, NC);
    {
        dim3 g(Mp / 128, 256 / 32);
        k_gemm<<<g, 256, 0, stream>>>(Hb, 128, Wl1t, 128, bl1, D1, 256, N, 128);
    }
    k_cvt<<<nb(N * 256), TB, 0, stream>>>(D1, 256, Hb, 256, 0, N * 256);
    {
        dim3 g(Mp / 128, 128 / 32);
        k_gemm<<<g, 256, 0, stream>>>(Hb, 256, Wl2t, 256, bl2, H1f, 128, N, 256);
    }
    k_cvt<<<nb(NC), TB, 0, stream>>>(H1f, 128, Hb, 128, 0, NC);
    {
        dim3 g(Mp / 128, 64 / 32);
        k_gemm<<<g, 256, 0, stream>>>(Hb, 128, Wot, 128, bo, (float*)d_out, 64, N, 128);
    }
}
